// ExpertGroup_60404420051197
// MI455X (gfx1250) — compile-verified, matrix operand order checked by reading, not run on hardware
//
#include <hip/hip_runtime.h>

// ---------------------------------------------------------------------------
// Grouped MoE MLP for MI455X (gfx1250, wave32, WMMA):
//   h   = relu(bf16(x) @ bf16(w_up[e])^T)^2        (bf16, fp32 accum)
//   out = f32(bf16( bf16(h) @ bf16(w_down[e])^T )) (fp32 accum, bf16 round)
// 128x128 block tile, BK=64, double-buffered LDS, v_cvt_pk_bf16_f32 staging,
// TDM (tensor_load_to_lds) for the bf16 intermediate in the down projection.
// ---------------------------------------------------------------------------

typedef __attribute__((ext_vector_type(16))) __bf16        v16bf;
typedef __attribute__((ext_vector_type(2)))  __bf16        v2bf;
typedef __attribute__((ext_vector_type(8)))  float         v8f;
typedef __attribute__((ext_vector_type(4)))  unsigned int  v4u;
typedef __attribute__((ext_vector_type(2)))  unsigned int  v2u;
typedef __attribute__((ext_vector_type(4)))  unsigned int  u32x4;
typedef __attribute__((ext_vector_type(8)))  int           i32x8;
typedef __attribute__((ext_vector_type(4)))  int           i32x4;

#define BM 128
#define BN 128
#define BK 64
#define LDSK (BK + 16)      // ushort stride -> 160B rows (16B aligned, padded)
#define NTHREADS 256

#define HAS_TDM __has_builtin(__builtin_amdgcn_tensor_load_to_lds)

union Frag { v16bf v; v4u q[2]; };

// ---- fp32 -> bf16 (RNE), packed: v_cvt_pk_bf16_f32 when available ---------
__device__ __forceinline__ unsigned pack_bf16x2(float a, float b) {
#if __has_builtin(__builtin_amdgcn_cvt_pk_bf16_f32)
  v2bf r = __builtin_amdgcn_cvt_pk_bf16_f32(a, b);
  return __builtin_bit_cast(unsigned, r);
#else
  unsigned ua = __float_as_uint(a); ua += 0x7FFFu + ((ua >> 16) & 1u);
  unsigned ub = __float_as_uint(b); ub += 0x7FFFu + ((ub >> 16) & 1u);
  return (ua >> 16) | (ub & 0xFFFF0000u);
#endif
}
__device__ __forceinline__ unsigned short f32_bf16(float f) {
  return (unsigned short)pack_bf16x2(f, f);
}
__device__ __forceinline__ float bf16_f32(unsigned short h) {
  return __uint_as_float(((unsigned int)h) << 16);
}

#if HAS_TDM
// 2D TDM: tile of 64 bf16 (dim0) x 128 rows (dim1), row stride K elements,
// into LDS with +32B padding after every 128B row (matches LDSK=80 ushorts).
__device__ __forceinline__ void tdm_load_tile_bf16(const unsigned short* g,
                                                   unsigned lds_off,
                                                   int K, int rows_left) {
  unsigned long long ga = (unsigned long long)(size_t)g;
  u32x4 g0;
  g0.x = 1u;                                       // count=1, user mode
  g0.y = lds_off;                                  // lds_addr
  g0.z = (unsigned)(ga & 0xFFFFFFFFull);           // global_addr[31:0]
  g0.w = (unsigned)((ga >> 32) & 0x1FFFFFFull)     // global_addr[56:32]
         | (2u << 30);                             // type = 2 ("image")
  i32x8 g1;
  g1[0] = (int)((1u << 16)                         // data_size = 2 bytes
              | (1u << 20)                         // pad_enable
              | (4u << 22)                         // pad_interval: 32 dwords
              | (7u << 25));                       // pad_amount:   8 dwords
  g1[1] = (int)(((unsigned)K & 0xFFFFu) << 16);    // tensor_dim0[15:0]
  g1[2] = (int)((((unsigned)K >> 16) & 0xFFFFu)    // tensor_dim0[31:16]
              | (((unsigned)rows_left & 0xFFFFu) << 16)); // tensor_dim1[15:0]
  g1[3] = (int)((((unsigned)rows_left >> 16) & 0xFFFFu)   // tensor_dim1[31:16]
              | (64u << 16));                      // tile_dim0 = 64
  g1[4] = 128;                                     // tile_dim1 = 128
  g1[5] = K;                                       // tensor_dim0_stride[31:0]
  g1[6] = 0;
  g1[7] = 0;
  i32x4 z4 = (i32x4)0;                             // groups 2/3 unused (2D)
  i32x8 z8 = (i32x8)0;
  __builtin_amdgcn_tensor_load_to_lds(g0, g1, z4, z4, z8, 0);
}
#endif

// A_F32: A operand fp32 in memory (convert while staging); else bf16 (TDM).
// SQRELU: bf16 relu^2 epilogue, bf16 store; else fp32 store rounded via bf16.
template <bool A_F32, bool SQRELU>
__global__ __launch_bounds__(NTHREADS)
void moe_gemm_wmma(const void* __restrict__ Aptr,
                   const float* __restrict__ W,
                   const int* __restrict__ counts, int E,
                   void* __restrict__ Cptr,
                   int K, int Ntot) {
  __shared__ __align__(16) unsigned short lA[2][BM * LDSK];
  __shared__ __align__(16) unsigned short lB[2][BN * LDSK];

  const int tid   = threadIdx.x;
  const int lane  = tid & 31;
  const int wid   = tid >> 5;
  const int waveM = (wid & 3) * 32;   // 4 waves along M
  const int waveN = (wid >> 2) * 64;  // 2 waves along N
  const int lo    = lane & 15;
  const int hi    = lane >> 4;

  const int row0 = blockIdx.y * BM;
  const int col0 = blockIdx.x * BN;

  // Uniform expert id for this row tile (segments are tile-aligned here).
  int e = E - 1;
  long long s = 0;
  for (int i = 0; i < E; ++i) {
    int c = counts[i];
    if (row0 < s + c) { e = i; break; }
    s += c;
  }
  const float* Wb = W + (size_t)e * (size_t)Ntot * (size_t)K;

  v8f acc[2][4];
#pragma unroll
  for (int m = 0; m < 2; ++m)
#pragma unroll
    for (int n = 0; n < 4; ++n) acc[m][n] = (v8f)0.0f;

  const int NK = K / BK;

  // ---- stage one BKxBM/BN tile pair into LDS buffer `buf` ----
  auto stage = [&](int kt, int buf) {
    const int k0 = kt * BK;
    if constexpr (A_F32) {
      const float* gA = (const float*)Aptr;
#pragma unroll
      for (int i = 0; i < 8; ++i) {
        int idx = tid + i * NTHREADS;
        int r = idx >> 4;
        int c = (idx & 15) << 2;
        const float4 f =
            *(const float4*)(gA + (size_t)(row0 + r) * K + k0 + c);
        v2u p;
        p.x = pack_bf16x2(f.x, f.y);
        p.y = pack_bf16x2(f.z, f.w);
        *(v2u*)&lA[buf][r * LDSK + c] = p;
      }
    } else {
#if HAS_TDM
      if (wid == 0) {  // TDM is per-wave; issue once per block
        const unsigned short* g =
            (const unsigned short*)Aptr + (size_t)row0 * K + k0;
        tdm_load_tile_bf16(g, (unsigned)(size_t)&lA[buf][0], K,
                           (int)(gridDim.y * BM) - row0);
      }
#else
      const unsigned short* gA = (const unsigned short*)Aptr;
#pragma unroll
      for (int i = 0; i < 4; ++i) {
        int idx = tid + i * NTHREADS;
        int r = idx >> 3;
        int c = (idx & 7) << 3;
        v4u d = *(const v4u*)(gA + (size_t)(row0 + r) * K + k0 + c);
        *(v4u*)&lA[buf][r * LDSK + c] = d;
      }
#endif
    }
    // B tile: fp32 weights, convert to bf16 while staging.
#pragma unroll
    for (int i = 0; i < 8; ++i) {
      int idx = tid + i * NTHREADS;
      int r = idx >> 4;
      int c = (idx & 15) << 2;
      const float4 f =
          *(const float4*)(Wb + (size_t)(col0 + r) * K + k0 + c);
      v2u p;
      p.x = pack_bf16x2(f.x, f.y);
      p.y = pack_bf16x2(f.z, f.w);
      *(v2u*)&lB[buf][r * LDSK + c] = p;
    }
    // Cache prefetch two tiles ahead on the weight stream.
    if (k0 + 2 * BK < K)
      __builtin_prefetch(Wb + (size_t)(col0 + (tid >> 1)) * K + k0 + 2 * BK,
                         0, 3);
  };

  // ---- 16x16x32 bf16 WMMA over one staged tile ----
  auto compute = [&](int buf) {
#pragma unroll
    for (int kk = 0; kk < BK; kk += 32) {
      // bf16 A/B ISA layout: lanes 0-15 hold K {kk..+7, kk+16..+23},
      // lanes 16-31 hold K {kk+8..+15, kk+24..+31}.
      const int kb = kk + hi * 8;
      Frag a[2], b[4];
#pragma unroll
      for (int m = 0; m < 2; ++m) {
        const v4u* p = (const v4u*)&lA[buf][(waveM + m * 16 + lo) * LDSK + kb];
        a[m].q[0] = p[0];
        a[m].q[1] = p[2];
      }
#pragma unroll
      for (int n = 0; n < 4; ++n) {
        const v4u* p = (const v4u*)&lB[buf][(waveN + n * 16 + lo) * LDSK + kb];
        b[n].q[0] = p[0];
        b[n].q[1] = p[2];
      }
#pragma unroll
      for (int m = 0; m < 2; ++m)
#pragma unroll
        for (int n = 0; n < 4; ++n)
          acc[m][n] = __builtin_amdgcn_wmma_f32_16x16x32_bf16(
              false, a[m].v, false, b[n].v, (short)0, acc[m][n],
              false, false);
    }
  };

  // ---- software pipeline: double-buffered LDS, one barrier per K tile ----
  stage(0, 0);
#if HAS_TDM
  if constexpr (!A_F32) {
    if (wid == 0) __builtin_amdgcn_s_wait_tensorcnt(0);
  }
#endif
  __syncthreads();
  for (int kt = 0; kt < NK; ++kt) {
    const int cur = kt & 1;
    if (kt + 1 < NK) stage(kt + 1, cur ^ 1);
    compute(cur);
#if HAS_TDM
    if constexpr (!A_F32) {
      if (wid == 0 && kt + 1 < NK) __builtin_amdgcn_s_wait_tensorcnt(0);
    }
#endif
    __syncthreads();
  }

  // ---- epilogue: C layout lanes 0-15 M=r, lanes 16-31 M=r+8, N=lo ----
#pragma unroll
  for (int m = 0; m < 2; ++m) {
#pragma unroll
    for (int n = 0; n < 4; ++n) {
      const int col  = col0 + waveN + n * 16 + lo;
      const int rowb = row0 + waveM + m * 16 + hi * 8;
#pragma unroll
      for (int r = 0; r < 8; ++r) {
        float v = acc[m][n][r];
        if (SQRELU) {
          // reference: bf16(h) -> relu -> square in bf16
          float hv = bf16_f32(f32_bf16(v));
          hv = hv > 0.0f ? hv : 0.0f;
          unsigned short us = f32_bf16(hv * hv);
          ((unsigned short*)Cptr)[(size_t)(rowb + r) * Ntot + col] = us;
        } else {
          // reference: bf16 matmul output upcast to fp32
          ((float*)Cptr)[(size_t)(rowb + r) * Ntot + col] =
              bf16_f32(f32_bf16(v));
        }
      }
    }
  }
}

extern "C" void kernel_launch(void* const* d_in, const int* in_sizes, int n_in,
                              void* d_out, int out_size, void* d_ws,
                              size_t ws_size, hipStream_t stream) {
  const float* x      = (const float*)d_in[0];  // (T, D) fp32
  const float* w_up   = (const float*)d_in[1];  // (E, H, D) fp32
  const float* w_down = (const float*)d_in[2];  // (E, D, H) fp32
  const int*   counts = (const int*)d_in[3];    // (E,) int32

  const int D = 1024;
  const int E = in_sizes[3];
  const int T = in_sizes[0] / D;
  const int H = in_sizes[1] / (E * D);

  unsigned short* hbuf = (unsigned short*)d_ws;  // (T, H) bf16 intermediate
  float* out = (float*)d_out;                    // (T, D) fp32

  dim3 blk(NTHREADS);

  // up: h = relu2(bf16(x) @ w_up[e]^T)      K = D, Ntot = H
  dim3 g1(H / BN, T / BM);
  moe_gemm_wmma<true, true>
      <<<g1, blk, 0, stream>>>(x, w_up, counts, E, hbuf, D, H);

  // down: out = bf16(h) @ w_down[e]^T       K = H, Ntot = D
  dim3 g2(D / BN, T / BM);
  moe_gemm_wmma<false, false>
      <<<g2, blk, 0, stream>>>(hbuf, w_down, counts, E, out, H, D);
}
